// Model_24910810317516
// MI455X (gfx1250) — compile-verified, matrix operand order checked by reading, not run on hardware
//
#include <hip/hip_runtime.h>
#include <cstdint>

// ---------------------------------------------------------------------------
// Types for CDNA5 WMMA (wave32, 16x16x32 bf16 -> f32)
// ---------------------------------------------------------------------------
typedef __attribute__((ext_vector_type(16))) __bf16 v16bf;
typedef __attribute__((ext_vector_type(8)))  float  v8f;

union FragB { v16bf v; unsigned u[8]; };

__device__ __forceinline__ unsigned short f32_bf16(float f) {
  unsigned u = __float_as_uint(f);
  unsigned r = (u + 0x7FFFu + ((u >> 16) & 1u)) >> 16;   // round-to-nearest-even
  return (unsigned short)r;
}
__device__ __forceinline__ float bf16_f32(unsigned short h) {
  return __uint_as_float(((unsigned)h) << 16);
}

// K position of dword v within a 16-bit A/B fragment (CDNA5 ISA 7.12.2):
// lanes 0-15 hold K {0..7,16..23}, lanes 16-31 hold K {8..15,24..31}, 2 per dword.
__device__ __forceinline__ int kbase(int v, int half) {
  return ((v < 4) ? (2 * v) : (16 + 2 * (v - 4))) + 8 * half;
}

// ---------------------------------------------------------------------------
// Prep: split x (B,N,4) -> xyz f32 (B,N,3) + feat bf16 (B,N,1)
// ---------------------------------------------------------------------------
__global__ void prep_kernel(const float* __restrict__ x, float* __restrict__ xyz,
                            unsigned short* __restrict__ feat, int total) {
  int i = blockIdx.x * blockDim.x + threadIdx.x;
  if (i >= total) return;
  xyz[i * 3 + 0] = x[i * 4 + 0];
  xyz[i * 3 + 1] = x[i * 4 + 1];
  xyz[i * 3 + 2] = x[i * 4 + 2];
  feat[i] = f32_bf16(x[i * 4 + 3]);
}

// ---------------------------------------------------------------------------
// Weight convert: W (K,N) f32 -> Wt (N,Kpad) bf16, zero-padded along K
// ---------------------------------------------------------------------------
__global__ void convert_wt_kernel(const float* __restrict__ W, unsigned short* __restrict__ wt,
                                  int K, int Kpad, int N) {
  int i = blockIdx.x * blockDim.x + threadIdx.x;
  int total = N * Kpad;
  if (i >= total) return;
  int n = i / Kpad, k = i - n * Kpad;
  float v = (k < K) ? W[(size_t)k * N + n] : 0.0f;
  wt[(size_t)n * Kpad + k] = f32_bf16(v);
}

// ---------------------------------------------------------------------------
// Farthest point sampling: one block per batch, dist kept in LDS
// dyn smem: dist[N] + red[256] + redi[256]
// ---------------------------------------------------------------------------
__global__ void fps_kernel(const float* __restrict__ xyz, int N, int npoint,
                           int* __restrict__ fidx) {
  extern __shared__ unsigned char dynsm[];
  float* dist = (float*)dynsm;
  float* red  = dist + N;
  int*   redi = (int*)(red + 256);
  __shared__ int s_last;

  int b = blockIdx.x;
  const float* p = xyz + (size_t)b * N * 3;
  for (int i = threadIdx.x; i < N; i += blockDim.x) dist[i] = 1e10f;
  if (threadIdx.x == 0) { s_last = 0; fidx[(size_t)b * npoint] = 0; }
  __syncthreads();

  for (int it = 1; it < npoint; ++it) {
    int last = s_last;
    float lx = p[last * 3 + 0], ly = p[last * 3 + 1], lz = p[last * 3 + 2];
    float best = -1.0f; int bi = 0;
    for (int i = threadIdx.x; i < N; i += blockDim.x) {
      float dx = p[i * 3 + 0] - lx, dy = p[i * 3 + 1] - ly, dz = p[i * 3 + 2] - lz;
      float d = dx * dx + dy * dy + dz * dz;
      float dd = fminf(dist[i], d);
      dist[i] = dd;
      if (dd > best) { best = dd; bi = i; }
    }
    red[threadIdx.x] = best; redi[threadIdx.x] = bi;
    __syncthreads();
    for (int s = 128; s > 0; s >>= 1) {
      if (threadIdx.x < s && red[threadIdx.x + s] > red[threadIdx.x]) {
        red[threadIdx.x]  = red[threadIdx.x + s];
        redi[threadIdx.x] = redi[threadIdx.x + s];
      }
      __syncthreads();
    }
    if (threadIdx.x == 0) { s_last = redi[0]; fidx[(size_t)b * npoint + it] = redi[0]; }
    __syncthreads();
  }
}

// ---------------------------------------------------------------------------
// Gather sampled centers
// ---------------------------------------------------------------------------
__global__ void gather_newxyz_kernel(const float* __restrict__ xyz, int N,
                                     const int* __restrict__ fidx, int S,
                                     float* __restrict__ nxyz, int total) {
  int i = blockIdx.x * blockDim.x + threadIdx.x;
  if (i >= total) return;
  int b = i / S;
  int pi = fidx[i];
  nxyz[(size_t)i * 3 + 0] = xyz[((size_t)b * N + pi) * 3 + 0];
  nxyz[(size_t)i * 3 + 1] = xyz[((size_t)b * N + pi) * 3 + 1];
  nxyz[(size_t)i * 3 + 2] = xyz[((size_t)b * N + pi) * 3 + 2];
}

// ---------------------------------------------------------------------------
// Ball query: first 32 smallest indices within radius, pad with first hit
// ---------------------------------------------------------------------------
__global__ void ball_query_kernel(const float* __restrict__ xyz, int N,
                                  const float* __restrict__ nxyz, int S, float r2,
                                  int* __restrict__ idx, int total) {
  int i = blockIdx.x * blockDim.x + threadIdx.x;
  if (i >= total) return;
  int b = i / S;
  const float* p = xyz + (size_t)b * N * 3;
  float cx = nxyz[(size_t)i * 3 + 0], cy = nxyz[(size_t)i * 3 + 1], cz = nxyz[(size_t)i * 3 + 2];
  int* o = idx + (size_t)i * 32;
  int cnt = 0;
  for (int j = 0; j < N && cnt < 32; ++j) {
    float dx = p[j * 3 + 0] - cx, dy = p[j * 3 + 1] - cy, dz = p[j * 3 + 2] - cz;
    if (dx * dx + dy * dy + dz * dz <= r2) o[cnt++] = j;
  }
  if (cnt == 0) {
    for (int t = 0; t < 32; ++t) o[t] = 0;
  } else {
    int f = o[0];
    for (int t = cnt; t < 32; ++t) o[t] = f;
  }
}

// ---------------------------------------------------------------------------
// WMMA tile layer: out(32,N) = relu(in(32,Kd) @ Wt^T + bias), all bf16 in LDS
// Wt stored as (N, Kd) so each lane's k-pair is one dword load.
// ---------------------------------------------------------------------------
__device__ void gemm_bias_relu(const unsigned short* __restrict__ in, int ldin, int Kd,
                               const unsigned short* __restrict__ wt,
                               const float* __restrict__ bias, int N,
                               unsigned short* __restrict__ out, int ldout) {
  const int wave = threadIdx.x >> 5;
  const int lane = threadIdx.x & 31;
  const int half = lane >> 4;
  const int lm   = lane & 15;
  const int nTiles = N >> 4;
  for (int nt = wave; nt < nTiles; nt += 4) {
    const int n = nt * 16 + lm;
    const unsigned short* wcol = wt + (size_t)n * Kd;
    v8f c0 = {}; v8f c1 = {};
    for (int k0 = 0; k0 < Kd; k0 += 32) {
      FragB a0, a1, bb;
#pragma unroll
      for (int v = 0; v < 8; ++v) {
        const int kb = kbase(v, half);
        a0.u[v] = *(const unsigned*)(&in[lm * ldin + k0 + kb]);
        a1.u[v] = *(const unsigned*)(&in[(lm + 16) * ldin + k0 + kb]);
        bb.u[v] = *(const unsigned*)(&wcol[k0 + kb]);
      }
      c0 = __builtin_amdgcn_wmma_f32_16x16x32_bf16(false, a0.v, false, bb.v, (short)0, c0, false, false);
      c1 = __builtin_amdgcn_wmma_f32_16x16x32_bf16(false, a1.v, false, bb.v, (short)0, c1, false, false);
    }
    const float bv = bias[n];
#pragma unroll
    for (int r = 0; r < 8; ++r) {
      float x0 = c0[r] + bv; x0 = x0 > 0.f ? x0 : 0.f;
      float x1 = c1[r] + bv; x1 = x1 > 0.f ? x1 : 0.f;
      const int m = r + 8 * half;
      out[(m)      * ldout + nt * 16 + lm] = f32_bf16(x0);
      out[(m + 16) * ldout + nt * 16 + lm] = f32_bf16(x1);
    }
  }
}

// Same, but fuses bias+relu+max over the 32 rows -> one output row per group.
__device__ void gemm_bias_relu_maxpool(const unsigned short* __restrict__ in, int ldin, int Kd,
                                       const unsigned short* __restrict__ wt,
                                       const float* __restrict__ bias, int N, int g,
                                       unsigned short* __restrict__ out_bf16,
                                       float* __restrict__ out_f32) {
  const int wave = threadIdx.x >> 5;
  const int lane = threadIdx.x & 31;
  const int half = lane >> 4;
  const int lm   = lane & 15;
  const int nTiles = N >> 4;
  for (int nt = wave; nt < nTiles; nt += 4) {
    const int n = nt * 16 + lm;
    const unsigned short* wcol = wt + (size_t)n * Kd;
    v8f c0 = {}; v8f c1 = {};
    for (int k0 = 0; k0 < Kd; k0 += 32) {
      FragB a0, a1, bb;
#pragma unroll
      for (int v = 0; v < 8; ++v) {
        const int kb = kbase(v, half);
        a0.u[v] = *(const unsigned*)(&in[lm * ldin + k0 + kb]);
        a1.u[v] = *(const unsigned*)(&in[(lm + 16) * ldin + k0 + kb]);
        bb.u[v] = *(const unsigned*)(&wcol[k0 + kb]);
      }
      c0 = __builtin_amdgcn_wmma_f32_16x16x32_bf16(false, a0.v, false, bb.v, (short)0, c0, false, false);
      c1 = __builtin_amdgcn_wmma_f32_16x16x32_bf16(false, a1.v, false, bb.v, (short)0, c1, false, false);
    }
    const float bv = bias[n];
    float mx = -3.0e38f;
#pragma unroll
    for (int r = 0; r < 8; ++r) {
      float x0 = c0[r] + bv; x0 = x0 > 0.f ? x0 : 0.f;
      float x1 = c1[r] + bv; x1 = x1 > 0.f ? x1 : 0.f;
      mx = fmaxf(mx, fmaxf(x0, x1));
    }
    mx = fmaxf(mx, __shfl_xor(mx, 16, 32));   // combine lane halves (m 0-7 with 8-15)
    if (half == 0) {
      const int col = nt * 16 + lm;
      if (out_bf16) out_bf16[(size_t)g * N + col] = f32_bf16(mx);
      if (out_f32)  out_f32 [(size_t)g * N + col] = mx;
    }
  }
}

// ---------------------------------------------------------------------------
// Fused per-group 3-layer MLP + maxpool.
// mode 0: gather via idx, recenter by new_xyz. mode 1: direct 32-row slice.
// dyn smem: bufA (32 x ldA bf16) + bufB (32 x ldB bf16)
// ---------------------------------------------------------------------------
__global__ void group_mlp_kernel(const float* __restrict__ xyz,
                                 const unsigned short* __restrict__ feat_in,
                                 int Cfeat, int Nin,
                                 const int* __restrict__ idx, const float* __restrict__ nxyz,
                                 int Sg, int mode,
                                 const unsigned short* __restrict__ w0t, const float* __restrict__ b0,
                                 int K0p, int N0,
                                 const unsigned short* __restrict__ w1t, const float* __restrict__ b1,
                                 int N1,
                                 const unsigned short* __restrict__ w2t, const float* __restrict__ b2,
                                 int N2,
                                 int ldA, int ldB,
                                 unsigned short* __restrict__ out_bf16, float* __restrict__ out_f32) {
  extern __shared__ unsigned char dynsm[];
  unsigned short* bufA = (unsigned short*)dynsm;
  unsigned short* bufB = bufA + 32 * ldA;

  const int g = blockIdx.x;
  const int b = g / Sg;
  const int sg = g - b * Sg;
  const int C0 = 3 + Cfeat;

  // Build input tile (32 x K0p), zero-padded K
  for (int i = threadIdx.x; i < 32 * K0p; i += blockDim.x) {
    int r = i / K0p, c = i - r * K0p;
    float val = 0.0f;
    int pidx;
    float cx = 0.f, cy = 0.f, cz = 0.f;
    if (mode == 0) {
      pidx = idx[(size_t)g * 32 + r];
      cx = nxyz[(size_t)g * 3 + 0];
      cy = nxyz[(size_t)g * 3 + 1];
      cz = nxyz[(size_t)g * 3 + 2];
    } else {
      pidx = sg * 32 + r;
    }
    if (c < 3) {
      float pv = xyz[((size_t)b * Nin + pidx) * 3 + c];
      float cv = (c == 0) ? cx : ((c == 1) ? cy : cz);
      val = pv - cv;
    } else if (c < C0) {
      val = bf16_f32(feat_in[((size_t)b * Nin + pidx) * Cfeat + (c - 3)]);
    }
    bufA[r * ldA + c] = f32_bf16(val);
  }
  __syncthreads();

  gemm_bias_relu(bufA, ldA, K0p, w0t, b0, N0, bufB, ldB);
  __syncthreads();
  gemm_bias_relu(bufB, ldB, N0, w1t, b1, N1, bufA, ldA);
  __syncthreads();
  gemm_bias_relu_maxpool(bufA, ldA, N1, w2t, b2, N2, g, out_bf16, out_f32);
}

// ---------------------------------------------------------------------------
// Final global max over the 4 row-groups of SA4
// ---------------------------------------------------------------------------
__global__ void reduce4_kernel(const float* __restrict__ partial, float* __restrict__ outf,
                               int total) {
  int i = blockIdx.x * blockDim.x + threadIdx.x;
  if (i >= total) return;
  int b = i / 1024, c = i - b * 1024;
  float m = partial[((size_t)b * 4 + 0) * 1024 + c];
  m = fmaxf(m, partial[((size_t)b * 4 + 1) * 1024 + c]);
  m = fmaxf(m, partial[((size_t)b * 4 + 2) * 1024 + c]);
  m = fmaxf(m, partial[((size_t)b * 4 + 3) * 1024 + c]);
  outf[i] = m;
}

// ---------------------------------------------------------------------------
// Heads: 8 rows only -> plain f32. out[0..55]=box, out[56..63]=sigmoid class.
// ---------------------------------------------------------------------------
__global__ void head_kernel(const float* __restrict__ feat,
                            const float* cw1, const float* cb1, const float* cw2,
                            const float* cb2, const float* cw3, const float* cb3,
                            const float* rw1, const float* rb1, const float* rw2,
                            const float* rb2, const float* rw3, const float* rb3,
                            float* __restrict__ out) {
  __shared__ float h0[8 * 256];
  __shared__ float h1[8 * 256];
  const int tid = threadIdx.x;
  for (int pass = 0; pass < 2; ++pass) {   // pass 0 = cls, pass 1 = reg
    const float* W1 = pass ? rw1 : cw1; const float* B1 = pass ? rb1 : cb1;
    const float* W2 = pass ? rw2 : cw2; const float* B2 = pass ? rb2 : cb2;
    const float* W3 = pass ? rw3 : cw3; const float* B3 = pass ? rb3 : cb3;
    for (int o = tid; o < 8 * 256; o += blockDim.x) {
      int r = o >> 8, c = o & 255;
      float acc = B1[c];
      for (int k = 0; k < 1024; ++k) acc += feat[r * 1024 + k] * W1[k * 256 + c];
      h0[o] = acc > 0.f ? acc : 0.f;
    }
    __syncthreads();
    for (int o = tid; o < 8 * 256; o += blockDim.x) {
      int r = o >> 8, c = o & 255;
      float acc = B2[c];
      for (int k = 0; k < 256; ++k) acc += h0[r * 256 + k] * W2[k * 256 + c];
      h1[o] = acc > 0.f ? acc : 0.f;
    }
    __syncthreads();
    const int Nout = pass ? 7 : 1;
    for (int o = tid; o < 8 * Nout; o += blockDim.x) {
      int r = o / Nout, c = o - r * Nout;
      float acc = B3[c];
      for (int k = 0; k < 256; ++k) acc += h1[r * 256 + k] * W3[k * Nout + c];
      if (pass) out[r * 7 + c] = acc;                       // pred_box
      else      out[56 + r]   = 1.f / (1.f + __expf(-acc)); // pred_class
    }
    __syncthreads();
  }
}

// ---------------------------------------------------------------------------
// Host launcher
// ---------------------------------------------------------------------------
extern "C" void kernel_launch(void* const* d_in, const int* in_sizes, int n_in,
                              void* d_out, int out_size, void* d_ws, size_t ws_size,
                              hipStream_t stream) {
  (void)in_sizes; (void)n_in; (void)out_size; (void)ws_size;

  const float* x = (const float*)d_in[0];
  const float* saW[4][3]; const float* saB[4][3];
  int p = 1;
  for (int i = 0; i < 4; ++i)
    for (int l = 0; l < 3; ++l) { saW[i][l] = (const float*)d_in[p++]; saB[i][l] = (const float*)d_in[p++]; }
  const float *cW[3], *cB[3], *rW[3], *rB[3];
  for (int l = 0; l < 3; ++l) { cW[l] = (const float*)d_in[p++]; cB[l] = (const float*)d_in[p++]; }
  for (int l = 0; l < 3; ++l) { rW[l] = (const float*)d_in[p++]; rB[l] = (const float*)d_in[p++]; }

  const int Bn = 8, Npts = 8192;
  static const int   S_[3]   = {2048, 512, 128};
  static const float R2_[3]  = {0.25f, 1.0f, 4.0f};
  static const int   Cin[4]  = {1, 128, 256, 512};
  static const int   C0p[4]  = {32, 160, 288, 544};       // (3+Cin) padded to 32
  static const int   Wd[4][3] = {{64, 64, 128}, {128, 128, 256}, {256, 256, 512}, {512, 512, 1024}};

  char* ws = (char*)d_ws;
  size_t off = 0;
  auto alloc = [&](size_t bytes) -> void* {
    void* r = ws + off;
    off += (bytes + 255) & ~(size_t)255;
    return r;
  };

  float*          xyz0  = (float*)alloc((size_t)Bn * Npts * 3 * 4);
  unsigned short* feat0 = (unsigned short*)alloc((size_t)Bn * Npts * 2);

  unsigned short* wt[4][3];
  for (int i = 0; i < 4; ++i)
    for (int l = 0; l < 3; ++l) {
      int Kp = (l == 0) ? C0p[i] : Wd[i][l - 1];
      wt[i][l] = (unsigned short*)alloc((size_t)Wd[i][l] * Kp * 2);
    }

  int*            fidx[3]; float* nxyz[3]; int* bidx[3]; unsigned short* featS[3];
  for (int i = 0; i < 3; ++i) {
    fidx[i]  = (int*)alloc((size_t)Bn * S_[i] * 4);
    nxyz[i]  = (float*)alloc((size_t)Bn * S_[i] * 3 * 4);
    bidx[i]  = (int*)alloc((size_t)Bn * S_[i] * 32 * 4);
    featS[i] = (unsigned short*)alloc((size_t)Bn * S_[i] * Wd[i][2] * 2);
  }
  float* partial = (float*)alloc((size_t)Bn * 4 * 1024 * 4);
  float* featF   = (float*)alloc((size_t)Bn * 1024 * 4);

  // Split input, convert weights once (bf16, transposed, K-padded)
  { int tot = Bn * Npts;
    prep_kernel<<<(tot + 255) / 256, 256, 0, stream>>>(x, xyz0, feat0, tot); }
  for (int i = 0; i < 4; ++i)
    for (int l = 0; l < 3; ++l) {
      int Kact = (l == 0) ? 3 + Cin[i] : Wd[i][l - 1];
      int Kp   = (l == 0) ? C0p[i]     : Wd[i][l - 1];
      int N    = Wd[i][l];
      int tot  = N * Kp;
      convert_wt_kernel<<<(tot + 255) / 256, 256, 0, stream>>>(saW[i][l], wt[i][l], Kact, Kp, N);
    }

  // Set-abstraction stages 1-3 (grouped)
  const float*          xyz_cur  = xyz0;
  const unsigned short* feat_cur = feat0;
  int Ncur = Npts;
  for (int i = 0; i < 3; ++i) {
    int S = S_[i];
    size_t shm_fps = ((size_t)Ncur + 512) * 4;
    fps_kernel<<<Bn, 256, shm_fps, stream>>>(xyz_cur, Ncur, S, fidx[i]);
    int tot = Bn * S;
    gather_newxyz_kernel<<<(tot + 255) / 256, 256, 0, stream>>>(xyz_cur, Ncur, fidx[i], S, nxyz[i], tot);
    ball_query_kernel<<<(tot + 255) / 256, 256, 0, stream>>>(xyz_cur, Ncur, nxyz[i], S, R2_[i], bidx[i], tot);

    int ldA = C0p[i] > Wd[i][1] ? C0p[i] : Wd[i][1];
    int ldB = Wd[i][0];
    size_t shm = (size_t)32 * (ldA + ldB) * 2;
    group_mlp_kernel<<<Bn * S, 128, shm, stream>>>(
        xyz_cur, feat_cur, Cin[i], Ncur, bidx[i], nxyz[i], S, 0,
        wt[i][0], saB[i][0], C0p[i], Wd[i][0],
        wt[i][1], saB[i][1], Wd[i][1],
        wt[i][2], saB[i][2], Wd[i][2],
        ldA, ldB, featS[i], (float*)nullptr);
    xyz_cur = nxyz[i]; feat_cur = featS[i]; Ncur = S;
  }

  // SA4: global stage, 4 row-groups of 32 per batch, then max-reduce
  {
    int ldA = 544, ldB = 512;
    size_t shm = (size_t)32 * (ldA + ldB) * 2;
    group_mlp_kernel<<<Bn * 4, 128, shm, stream>>>(
        xyz_cur, feat_cur, 512, 128, (const int*)nullptr, (const float*)nullptr, 4, 1,
        wt[3][0], saB[3][0], 544, 512,
        wt[3][1], saB[3][1], 512,
        wt[3][2], saB[3][2], 1024,
        ldA, ldB, (unsigned short*)nullptr, partial);
    int tot = Bn * 1024;
    reduce4_kernel<<<(tot + 255) / 256, 256, 0, stream>>>(partial, featF, tot);
  }

  // Heads
  head_kernel<<<1, 256, 0, stream>>>(featF, cW[0], cB[0], cW[1], cB[1], cW[2], cB[2],
                                     rW[0], rB[0], rW[1], rB[1], rW[2], rB[2], (float*)d_out);
}
